// DynamicGAT_89472758710442
// MI455X (gfx1250) — matrix operand order repaired and checked by reading, not compiled, not run on hardware
//
#include <hip/hip_runtime.h>
#include <hip/hip_bf16.h>
#include <math.h>

typedef __attribute__((ext_vector_type(16))) _Float16 v16h;
typedef __attribute__((ext_vector_type(8)))  _Float16 v8h;
typedef __attribute__((ext_vector_type(8)))  float    v8f;

static constexpr int NN   = 2048;
static constexpr int INF_ = 64;
static constexpr int DD   = 128;
static constexpr int HH   = 4;
static constexpr int DHH  = 32;
static constexpr int KTOP = 64;
static constexpr int LLAY = 2;
static constexpr int E00  = 32768;
#define THRESHV 0.5f
#define TEMPV   0.5f
#define SLOPEV  0.2f

// ---------- helpers ----------
__device__ __forceinline__ unsigned int fkey(float x) {
  unsigned int u = __float_as_uint(x);
  return (u & 0x80000000u) ? ~u : (u | 0x80000000u);
}
__device__ __forceinline__ float keyf(unsigned int k) {
  unsigned int u = (k & 0x80000000u) ? (k & 0x7fffffffu) : ~k;
  return __uint_as_float(u);
}
__device__ __forceinline__ float lrelu(float x) { return x > 0.0f ? x : SLOPEV * x; }

// Convert two float4 (8 consecutive floats) into 8 packed halfs.
__device__ __forceinline__ v8h cvt8(const float4 f0, const float4 f1) {
  v8h r;
  r[0] = (_Float16)f0.x; r[1] = (_Float16)f0.y; r[2] = (_Float16)f0.z; r[3] = (_Float16)f0.w;
  r[4] = (_Float16)f1.x; r[5] = (_Float16)f1.y; r[6] = (_Float16)f1.z; r[7] = (_Float16)f1.w;
  return r;
}

// Build an A-fragment (16x32 f16, contiguous k order) directly from a global
// row-major row: lane's 16 halfs are row[kb..kb+15],
// kb = kt + 16*(lane>=16).  Requires 16B alignment (K stride multiple of 8).
__device__ __forceinline__ v16h afrag_global(const float* __restrict__ row, int kt, int lane) {
  int kb = kt + ((lane >= 16) ? 16 : 0);
  const float4* p = (const float4*)(row + kb);
  float4 f0 = p[0], f1 = p[1], f2 = p[2], f3 = p[3];
  v16h a;
  a[0]=(_Float16)f0.x; a[1]=(_Float16)f0.y; a[2]=(_Float16)f0.z; a[3]=(_Float16)f0.w;
  a[4]=(_Float16)f1.x; a[5]=(_Float16)f1.y; a[6]=(_Float16)f1.z; a[7]=(_Float16)f1.w;
  a[8]=(_Float16)f2.x; a[9]=(_Float16)f2.y; a[10]=(_Float16)f2.z; a[11]=(_Float16)f2.w;
  a[12]=(_Float16)f3.x; a[13]=(_Float16)f3.y; a[14]=(_Float16)f3.z; a[15]=(_Float16)f3.w;
  return a;
}

// ---------- generic WMMA GEMM: C = act(A[MxK] @ B[KxN] (+ Cadd)) ----------
// Block 128 threads (4 waves), 64x64 tile. A-frags direct from global;
// B staged in LDS in fragment order (4 subtiles of 512 halfs).
__global__ __launch_bounds__(128)
void k_gemm(const float* __restrict__ A, const float* __restrict__ B,
            const float* __restrict__ Cadd, float* __restrict__ C,
            int M, int Nn, int Kk, int act) {
  __shared__ alignas(32) _Float16 BsF[4 * 512];
  const int tid = threadIdx.x, lane = tid & 31, wv = tid >> 5;
  const int m0 = blockIdx.y * 64, n0 = blockIdx.x * 64;
  v8f vzero = {0.f,0.f,0.f,0.f,0.f,0.f,0.f,0.f};
  v8f acc[4];
  acc[0] = vzero; acc[1] = vzero; acc[2] = vzero; acc[3] = vzero;
  const int arow = m0 + wv * 16 + (lane & 15);
  const float* Arow = A + (size_t)((arow < M) ? arow : 0) * Kk;

  for (int kt = 0; kt < Kk; kt += 32) {
    // stage B[kt..kt+32) x [n0..n0+64) into fragment order
    for (int u = tid; u < 256; u += 128) {
      int kg = u >> 6, c = u & 63;             // K-group of 8, column
      v8h hv;
#pragma unroll
      for (int i = 0; i < 8; ++i)
        hv[i] = (_Float16)B[(size_t)(kt + kg * 8 + i) * Nn + n0 + c];
      int L = (c & 15) + ((kg >= 2) ? 16 : 0);
      int dst = (c >> 4) * 512 + L * 16 + (kg & 1) * 8;
      *(v8h*)(BsF + dst) = hv;
    }
    __syncthreads();
    v16h af = afrag_global(Arow, kt, lane);
#pragma unroll
    for (int s = 0; s < 4; ++s) {
      v16h bf = *(const v16h*)(BsF + s * 512 + lane * 16);
      acc[s] = __builtin_amdgcn_wmma_f32_16x16x32_f16(false, af, false, bf,
                                                      (short)0, acc[s], false, false);
    }
    __syncthreads();
  }
#pragma unroll
  for (int s = 0; s < 4; ++s) {
#pragma unroll
    for (int vv = 0; vv < 8; ++vv) {
      int row = m0 + wv * 16 + vv + ((lane >= 16) ? 8 : 0);
      int col = n0 + s * 16 + (lane & 15);
      if (row < M) {
        float val = acc[s][vv];
        if (Cadd) val += Cadd[(size_t)row * Nn + col];
        if (act) val = lrelu(val);
        C[(size_t)row * Nn + col] = val;
      }
    }
  }
}

// ---------- attention scores: S[h][n][m] = (q_nh . k_mh)/sqrt(32) ----------
// K = DH = 32: one WMMA k-step. A-frag direct from q; k^T staged frag-order.
__global__ __launch_bounds__(128)
void k_scores(const float* __restrict__ Q, const float* __restrict__ Kb,
              float* __restrict__ S) {
  __shared__ alignas(32) _Float16 BsF[4 * 512];
  const int tid = threadIdx.x, lane = tid & 31, wv = tid >> 5;
  const int n0 = blockIdx.y * 64;
  const int m0 = blockIdx.x * 64;
  const int h  = blockIdx.z;
  // stage B: Btile[kk][c] = Kb[(m0+c)*DD + h*32 + kk]; contiguous in kk -> float4 loads
  for (int u = tid; u < 256; u += 128) {
    int kg = u >> 6, c = u & 63;
    const float4* p = (const float4*)(Kb + (size_t)(m0 + c) * DD + h * DHH + kg * 8);
    v8h hv = cvt8(p[0], p[1]);
    int L = (c & 15) + ((kg >= 2) ? 16 : 0);
    int dst = (c >> 4) * 512 + L * 16 + (kg & 1) * 8;
    *(v8h*)(BsF + dst) = hv;
  }
  __syncthreads();
  const float* qrow = Q + (size_t)(n0 + wv * 16 + (lane & 15)) * DD + h * DHH;
  v16h af = afrag_global(qrow, 0, lane);
  v8f vzero = {0.f,0.f,0.f,0.f,0.f,0.f,0.f,0.f};
#pragma unroll
  for (int s = 0; s < 4; ++s) {
    v16h bf = *(const v16h*)(BsF + s * 512 + lane * 16);
    v8f acc = __builtin_amdgcn_wmma_f32_16x16x32_f16(false, af, false, bf,
                                                     (short)0, vzero, false, false);
#pragma unroll
    for (int vv = 0; vv < 8; ++vv) {
      int row = n0 + wv * 16 + vv + ((lane >= 16) ? 8 : 0);
      int col = m0 + s * 16 + (lane & 15);
      S[((size_t)h * NN + row) * NN + col] = acc[vv] * 0.17677669529663689f;
    }
  }
}

// ---------- per-row: topK->softmax->head-sum->sigmoid->entropy->dynamic edges ----------
__global__ __launch_bounds__(256)
void k_row(const float* __restrict__ S, float* __restrict__ probs,
           float* __restrict__ penalty,
           int* __restrict__ esrc, int* __restrict__ edst,
           float* __restrict__ ew, float* __restrict__ eev, int Eprev) {
  __shared__ float srow[NN];
  __shared__ float asum[NN];
  __shared__ float red[256];
  __shared__ int   scnt;
  const int n = blockIdx.x;
  const int tid = threadIdx.x;
  for (int m = tid; m < NN; m += 256) asum[m] = 0.0f;

  for (int h = 0; h < HH; ++h) {
    __syncthreads();
    for (int m = tid; m < NN; m += 256)
      srow[m] = S[((size_t)h * NN + n) * NN + m];
    __syncthreads();
    float lmax = -3.4e38f;
    for (int m = tid; m < NN; m += 256) lmax = fmaxf(lmax, srow[m]);
    red[tid] = lmax; __syncthreads();
    for (int s = 128; s > 0; s >>= 1) { if (tid < s) red[tid] = fmaxf(red[tid], red[tid + s]); __syncthreads(); }
    float rmax = red[0]; __syncthreads();
    // radix-select K-th largest (exact)
    unsigned int thr = 0u;
    for (int bit = 31; bit >= 0; --bit) {
      unsigned int cand = thr | (1u << bit);
      int lc = 0;
      for (int m = tid; m < NN; m += 256) lc += (fkey(srow[m]) >= cand) ? 1 : 0;
      red[tid] = (float)lc; __syncthreads();
      for (int s = 128; s > 0; s >>= 1) { if (tid < s) red[tid] += red[tid + s]; __syncthreads(); }
      if ((int)red[0] >= KTOP) thr = cand;
      __syncthreads();
    }
    float kth = keyf(thr);
    float lsum = 0.0f;
    for (int m = tid; m < NN; m += 256) { float x = srow[m]; if (x >= kth) lsum += __expf(x - rmax); }
    red[tid] = lsum; __syncthreads();
    for (int s = 128; s > 0; s >>= 1) { if (tid < s) red[tid] += red[tid + s]; __syncthreads(); }
    float denom = red[0]; __syncthreads();
    for (int m = tid; m < NN; m += 256) {
      float x = srow[m];
      if (x >= kth) asum[m] += __expf(x - rmax) / denom;
    }
  }
  __syncthreads();
  float lent = 0.0f;
  for (int m = tid; m < NN; m += 256) {
    float p = 1.0f / (1.0f + __expf(-asum[m] / TEMPV));
    probs[(size_t)n * NN + m] = p;
    srow[m] = p;
    lent += -p * __logf(p + 1e-10f);
  }
  red[tid] = lent; __syncthreads();
  for (int s = 128; s > 0; s >>= 1) { if (tid < s) red[tid] += red[tid + s]; __syncthreads(); }
  if (tid == 0) atomicAdd(penalty, red[0]);
  __syncthreads();
  if (tid == 0) {
    int idx = Eprev + NN * KTOP + n;
    esrc[idx] = n; edst[idx] = n; ew[idx] = srow[n]; eev[idx] = 1.0f;
  }
  unsigned int thr2 = 0u;
  for (int bit = 31; bit >= 0; --bit) {
    unsigned int cand = thr2 | (1u << bit);
    int lc = 0;
    for (int m = tid; m < NN; m += 256) lc += (fkey(srow[m]) >= cand) ? 1 : 0;
    red[tid] = (float)lc; __syncthreads();
    for (int s = 128; s > 0; s >>= 1) { if (tid < s) red[tid] += red[tid + s]; __syncthreads(); }
    if ((int)red[0] >= KTOP) thr2 = cand;
    __syncthreads();
  }
  if (tid == 0) scnt = 0;
  __syncthreads();
  for (int m = tid; m < NN; m += 256) {
    if (fkey(srow[m]) > thr2) {
      int pos = atomicAdd(&scnt, 1);
      int idx = Eprev + n * KTOP + pos;
      esrc[idx] = n; edst[idx] = m; ew[idx] = srow[m];
      eev[idx] = (srow[m] > THRESHV) ? 1.0f : 0.0f;
    }
  }
  __syncthreads();
  if (tid == 0) {
    int j = scnt;
    for (int m = 0; m < NN && j < KTOP; ++m) {
      if (fkey(srow[m]) == thr2) {
        int idx = Eprev + n * KTOP + j;
        esrc[idx] = n; edst[idx] = m; ew[idx] = srow[m];
        eev[idx] = (srow[m] > THRESHV) ? 1.0f : 0.0f;
        ++j;
      }
    }
  }
}

// ---------- refresh old-edge weights from this layer's probs ----------
__global__ void k_gatherw(const float* __restrict__ probs,
                          const int* __restrict__ esrc, const int* __restrict__ edst,
                          float* __restrict__ ew, int Eprev) {
  int e = blockIdx.x * 256 + threadIdx.x;
  if (e < Eprev) ew[e] = probs[(size_t)esrc[e] * NN + edst[e]];
}

__global__ void k_initedges(const int* __restrict__ ei, int* __restrict__ esrc,
                            int* __restrict__ edst, float* __restrict__ eev) {
  int e = blockIdx.x * 256 + threadIdx.x;
  if (e < E00) { esrc[e] = ei[e]; edst[e] = ei[E00 + e]; eev[e] = 1.0f; }
}

// ---------- batched-64-edge WMMA edge-MLP + scatter-add ----------
// msg_e = lrelu(Ad[dst_e]+Bsr[src_e]) @ W2 * (w_e*vmask_e); agg[dst_e]+=msg; cnt[dst_e]+=vmask
// Ts (gathered activations) and W2 staged in LDS in fragment order.
__global__ __launch_bounds__(256)
void k_edge(const float* __restrict__ Ad, const float* __restrict__ Bsr,
            const float* __restrict__ W2,
            const int* __restrict__ esrc, const int* __restrict__ edst,
            const float* __restrict__ ew, const float* __restrict__ eev,
            const unsigned char* __restrict__ mask, int E,
            float* __restrict__ agg, float* __restrict__ cnt) {
  __shared__ alignas(32) _Float16 W2s[32 * 512];  // 32 KB, frag order: (ks*8+cs)*512
  __shared__ alignas(32) _Float16 Ts[16 * 512];   // 16 KB, frag order: (eg*4+ks)*512
  __shared__ float sc[64];
  __shared__ int   dsts[64];
  __shared__ unsigned char vld[64];
  const int tid = threadIdx.x, lane = tid & 31, wv = tid >> 5;
  const int e0 = blockIdx.x * 64;
  // stage W2 (128x128) in fragment order; per-i loads coalesced across c
  for (int u = tid; u < 2048; u += 256) {
    int kg = u >> 7, c = u & 127;             // kg: K-group of 8 (0..15)
    v8h hv;
#pragma unroll
    for (int i = 0; i < 8; ++i)
      hv[i] = (_Float16)W2[(size_t)(kg * 8 + i) * DD + c];
    int L = (c & 15) + (((kg & 3) >= 2) ? 16 : 0);
    int dst = ((kg >> 2) * 8 + (c >> 4)) * 512 + L * 16 + (kg & 1) * 8;
    *(v8h*)(W2s + dst) = hv;
  }
  if (tid < 64) {
    int ge = e0 + tid;
    if (ge < E) {
      float vm = eev[ge];
      if (mask) vm *= (mask[esrc[ge]] ? 1.0f : 0.0f) * (mask[edst[ge]] ? 1.0f : 0.0f);
      sc[tid] = ew[ge] * vm;
      dsts[tid] = edst[ge];
      vld[tid] = 1;
      if (vm != 0.0f) atomicAdd(&cnt[edst[ge]], vm);
    } else { sc[tid] = 0.0f; dsts[tid] = 0; vld[tid] = 0; }
  }
  // gather lrelu(Ad[dst]+Bsr[src]) into fragment order (float4 pair loads)
  for (int u = tid; u < 1024; u += 256) {
    int le = u >> 4, g = u & 15;              // local edge, K-group of 8
    int ge = e0 + le;
    v8h hv;
    if (ge < E) {
      int d = edst[ge], s = esrc[ge];
      const float4* pa = (const float4*)(Ad + (size_t)d * DD + g * 8);
      const float4* pb = (const float4*)(Bsr + (size_t)s * DD + g * 8);
      float4 a0 = pa[0], a1 = pa[1], b0 = pb[0], b1 = pb[1];
      hv[0] = (_Float16)lrelu(a0.x + b0.x); hv[1] = (_Float16)lrelu(a0.y + b0.y);
      hv[2] = (_Float16)lrelu(a0.z + b0.z); hv[3] = (_Float16)lrelu(a0.w + b0.w);
      hv[4] = (_Float16)lrelu(a1.x + b1.x); hv[5] = (_Float16)lrelu(a1.y + b1.y);
      hv[6] = (_Float16)lrelu(a1.z + b1.z); hv[7] = (_Float16)lrelu(a1.w + b1.w);
    } else {
      v8h z = {};
      hv = z;
    }
    int rr = le & 15, eg = le >> 4, ks = g >> 2, gl = g & 3;
    int L = rr + ((gl >= 2) ? 16 : 0);
    int dst = (eg * 4 + ks) * 512 + L * 16 + (gl & 1) * 8;
    *(v8h*)(Ts + dst) = hv;
  }
  __syncthreads();
  v8f vzero = {0.f,0.f,0.f,0.f,0.f,0.f,0.f,0.f};
#pragma unroll
  for (int ti = 0; ti < 4; ++ti) {
    int t  = wv * 4 + ti;
    int eg = t >> 3;   // edge group of 16
    int cs = t & 7;    // 16-wide column strip
    v8f acc = vzero;
#pragma unroll
    for (int ks = 0; ks < 4; ++ks) {
      v16h af = *(const v16h*)(Ts + (eg * 4 + ks) * 512 + lane * 16);
      v16h bf = *(const v16h*)(W2s + (ks * 8 + cs) * 512 + lane * 16);
      acc = __builtin_amdgcn_wmma_f32_16x16x32_f16(false, af, false, bf,
                                                   (short)0, acc, false, false);
    }
#pragma unroll
    for (int vv = 0; vv < 8; ++vv) {
      int le  = eg * 16 + vv + ((lane >= 16) ? 8 : 0);
      int col = cs * 16 + (lane & 15);
      if (vld[le] && sc[le] != 0.0f)
        atomicAdd(&agg[(size_t)dsts[le] * DD + col], acc[vv] * sc[le]);
    }
  }
}

__global__ void k_div(float* __restrict__ agg, const float* __restrict__ cnt, int n) {
  int i = blockIdx.x * 256 + threadIdx.x;
  if (i < n) agg[i] /= fmaxf(cnt[i / DD], 1.0f);
}

// ---------- LayerNorm, one wave per row ----------
__global__ __launch_bounds__(256)
void k_ln(const float* __restrict__ X, const float* __restrict__ g,
          const float* __restrict__ b, float* __restrict__ Y) {
  int row = blockIdx.x * 8 + (threadIdx.x >> 5);
  int lane = threadIdx.x & 31;
  if (row >= NN) return;
  const float* x = X + (size_t)row * DD;
  float vals[4];
  float s = 0.0f;
#pragma unroll
  for (int i = 0; i < 4; ++i) { vals[i] = x[lane + i * 32]; s += vals[i]; }
  for (int o = 16; o > 0; o >>= 1) s += __shfl_xor(s, o, 32);
  float mu = s * (1.0f / DD);
  float v = 0.0f;
#pragma unroll
  for (int i = 0; i < 4; ++i) { float d = vals[i] - mu; v += d * d; }
  for (int o = 16; o > 0; o >>= 1) v += __shfl_xor(v, o, 32);
  float rs = rsqrtf(v * (1.0f / DD) + 1e-5f);
#pragma unroll
  for (int i = 0; i < 4; ++i) {
    int c = lane + i * 32;
    Y[(size_t)row * DD + c] = (vals[i] - mu) * rs * g[c] + b[c];
  }
}

// ---------- mean pool + reg/cls heads + penalty writeout ----------
__global__ __launch_bounds__(128)
void k_heads(const float* __restrict__ hbuf, const int* __restrict__ batch,
             const float* __restrict__ rW1, const float* __restrict__ rb1,
             const float* __restrict__ rW2, const float* __restrict__ rb2,
             const float* __restrict__ cW1, const float* __restrict__ cb1,
             const float* __restrict__ cW2, const float* __restrict__ cb2,
             const float* __restrict__ penalty, float* __restrict__ out) {
  __shared__ float pooled[DD];
  __shared__ float hr[64], hc[64];
  int tid = threadIdx.x;
  float s = 0.0f, c = 0.0f;
  for (int n = 0; n < NN; ++n) {
    if (batch[n] == 0) { s += hbuf[(size_t)n * DD + tid]; c += 1.0f; }
  }
  pooled[tid] = s / fmaxf(c, 1.0f);
  __syncthreads();
  if (tid < 64) {
    float a = rb1[tid], bb = cb1[tid];
    for (int k = 0; k < DD; ++k) { float p = pooled[k]; a += p * rW1[k * 64 + tid]; bb += p * cW1[k * 64 + tid]; }
    hr[tid] = lrelu(a);
    hc[tid] = lrelu(bb);
  }
  __syncthreads();
  if (tid == 0) {
    float r = rb2[0];
    for (int k = 0; k < 64; ++k) r += hr[k] * rW2[k];
    out[(size_t)NN * DD] = r;
    out[(size_t)NN * DD + 5] = penalty[0];
  }
  if (tid < 4) {
    float cc = cb2[tid];
    for (int k = 0; k < 64; ++k) cc += hc[k] * cW2[k * 4 + tid];
    out[(size_t)NN * DD + 1 + tid] = cc;
  }
}

// ---------- host-side launcher ----------
extern "C" void kernel_launch(void* const* d_in, const int* in_sizes, int n_in,
                              void* d_out, int out_size, void* d_ws, size_t ws_size,
                              hipStream_t stream) {
  (void)in_sizes; (void)n_in; (void)out_size; (void)ws_size;
  const float* x       = (const float*)d_in[0];
  const int*   ei      = (const int*)d_in[1];
  const int*   batch   = (const int*)d_in[2];
  const unsigned char* mask = (const unsigned char*)d_in[3];
  const float* emb_W   = (const float*)d_in[4];
  const float* dec_tab = (const float*)d_in[5];
  const float* Wq      = (const float*)d_in[6];
  const float* Wk      = (const float*)d_in[7];
  const float* Wn1     = (const float*)d_in[8];
  const float* Wn2     = (const float*)d_in[9];
  const float* Wa1     = (const float*)d_in[10];
  const float* Wa2     = (const float*)d_in[11];
  const float* Wt1     = (const float*)d_in[12];
  const float* Wt2     = (const float*)d_in[13];
  const float* Wm1     = (const float*)d_in[14];
  const float* Wm2     = (const float*)d_in[15];
  const float* ln_vg   = (const float*)d_in[16];
  const float* ln_vb   = (const float*)d_in[17];
  const float* ln_ag   = (const float*)d_in[18];
  const float* ln_ab   = (const float*)d_in[19];
  const float* skip_W  = (const float*)d_in[20];
  const float* reg_W1  = (const float*)d_in[21];
  const float* reg_b1  = (const float*)d_in[22];
  const float* reg_W2  = (const float*)d_in[23];
  const float* reg_b2  = (const float*)d_in[24];
  const float* cls_W1  = (const float*)d_in[25];
  const float* cls_b1  = (const float*)d_in[26];
  const float* cls_W2  = (const float*)d_in[27];
  const float* cls_b2  = (const float*)d_in[28];

  float* ws = (float*)d_ws;
  size_t off = 0;
  auto alloc = [&](size_t nelem) { float* p = ws + off; off += nelem; return p; };
  float* S      = alloc((size_t)HH * NN * NN);   // 67 MB
  float* probs  = alloc((size_t)NN * NN);        // 16.8 MB
  float* qb     = alloc((size_t)NN * DD);
  float* kb     = alloc((size_t)NN * DD);
  float* hbuf   = alloc((size_t)NN * DD);
  float* xinit  = alloc((size_t)NN * DD);
  float* decb   = alloc((size_t)NN * DD);
  float* An     = alloc((size_t)NN * DD);
  float* Bn     = alloc((size_t)NN * DD);
  float* Aa     = alloc((size_t)NN * DD);
  float* Ba     = alloc((size_t)NN * DD);
  float* agg    = alloc((size_t)NN * DD);
  float* aagg   = alloc((size_t)NN * DD);
  float* tmp    = alloc((size_t)NN * DD);
  float* outb   = alloc((size_t)NN * DD);
  float* attout = alloc((size_t)NN * DD);
  float* cnt    = alloc(NN);
  float* acnt   = alloc(NN);
  float* pen    = alloc(16);
  const int EMAX = E00 + LLAY * (NN * KTOP + NN);  // 299008
  int*   esrc = (int*)alloc(EMAX);
  int*   edst = (int*)alloc(EMAX);
  float* ew   = alloc(EMAX);
  float* eev  = alloc(EMAX);

  dim3 ggrid(DD / 64, NN / 64);

  // embed + init
  k_gemm<<<ggrid, 128, 0, stream>>>(x, emb_W, nullptr, hbuf, NN, DD, INF_, 0);
  hipMemcpyAsync(xinit, hbuf, (size_t)NN * DD * 4, hipMemcpyDeviceToDevice, stream);
  hipMemcpyAsync(decb, dec_tab, (size_t)NN * DD * 4, hipMemcpyDeviceToDevice, stream);
  k_initedges<<<(E00 + 255) / 256, 256, 0, stream>>>(ei, esrc, edst, eev);
  hipMemsetAsync(pen, 0, 4, stream);

  int Eprev = E00;
  for (int l = 0; l < LLAY; ++l) {
    const size_t w1off = (size_t)l * 2 * DD * DD;   // [2D, D] weights
    const size_t w2off = (size_t)l * DD * DD;       // [D, D] weights
    // q/k projections
    k_gemm<<<ggrid, 128, 0, stream>>>(decb, Wq + w2off, nullptr, qb, NN, DD, DD, 0);
    k_gemm<<<ggrid, 128, 0, stream>>>(decb, Wk + w2off, nullptr, kb, NN, DD, DD, 0);
    // per-head attention scores (WMMA)
    dim3 sg(NN / 64, NN / 64, HH);
    k_scores<<<sg, 128, 0, stream>>>(qb, kb, S);
    // top-K softmax / sigmoid / entropy / dynamic edges
    k_row<<<NN, 256, 0, stream>>>(S, probs, pen, esrc, edst, ew, eev, Eprev);
    k_gatherw<<<(Eprev + 255) / 256, 256, 0, stream>>>(probs, esrc, edst, ew, Eprev);
    int Enow = Eprev + NN * KTOP + NN;
    // node-level projections for both edge MLPs
    k_gemm<<<ggrid, 128, 0, stream>>>(hbuf, Wn1 + w1off,              nullptr, An, NN, DD, DD, 0);
    k_gemm<<<ggrid, 128, 0, stream>>>(hbuf, Wn1 + w1off + DD * DD,    nullptr, Bn, NN, DD, DD, 0);
    k_gemm<<<ggrid, 128, 0, stream>>>(decb, Wa1 + w1off,              nullptr, Aa, NN, DD, DD, 0);
    k_gemm<<<ggrid, 128, 0, stream>>>(decb, Wa1 + w1off + DD * DD,    nullptr, Ba, NN, DD, DD, 0);
    hipMemsetAsync(agg,  0, (size_t)NN * DD * 4, stream);
    hipMemsetAsync(aagg, 0, (size_t)NN * DD * 4, stream);
    hipMemsetAsync(cnt,  0, NN * 4, stream);
    hipMemsetAsync(acnt, 0, NN * 4, stream);
    int eb = (Enow + 63) / 64;
    k_edge<<<eb, 256, 0, stream>>>(An, Bn, Wn2 + w2off, esrc, edst, ew, eev, mask,    Enow, agg,  cnt);
    k_edge<<<eb, 256, 0, stream>>>(Aa, Ba, Wa2 + w2off, esrc, edst, ew, eev, nullptr, Enow, aagg, acnt);
    k_div<<<(NN * DD + 255) / 256, 256, 0, stream>>>(agg,  cnt,  NN * DD);
    k_div<<<(NN * DD + 255) / 256, 256, 0, stream>>>(aagg, acnt, NN * DD);
    // val update MLP + residual skip + LN -> h
    k_gemm<<<ggrid, 128, 0, stream>>>(agg,  Wt1 + w1off,           nullptr, tmp, NN, DD, DD, 0);
    k_gemm<<<ggrid, 128, 0, stream>>>(hbuf, Wt1 + w1off + DD * DD, tmp,     tmp, NN, DD, DD, 1);
    k_gemm<<<ggrid, 128, 0, stream>>>(tmp,  Wt2 + w2off,           nullptr, outb, NN, DD, DD, 0);
    k_gemm<<<ggrid, 128, 0, stream>>>(xinit, skip_W + w2off,       outb,    outb, NN, DD, DD, 0);
    k_ln<<<NN / 8, 256, 0, stream>>>(outb, ln_vg + l * DD, ln_vb + l * DD, hbuf);
    // att update MLP + LN -> dec
    k_gemm<<<ggrid, 128, 0, stream>>>(aagg, Wm1 + w1off,           nullptr, tmp, NN, DD, DD, 0);
    k_gemm<<<ggrid, 128, 0, stream>>>(decb, Wm1 + w1off + DD * DD, tmp,     tmp, NN, DD, DD, 1);
    k_gemm<<<ggrid, 128, 0, stream>>>(tmp,  Wm2 + w2off,           nullptr, attout, NN, DD, DD, 0);
    k_ln<<<NN / 8, 256, 0, stream>>>(attout, ln_ag + l * DD, ln_ab + l * DD, decb);
    Eprev = Enow;
  }

  // outputs: [h | reg | cls(4) | penalty]
  hipMemcpyAsync(d_out, hbuf, (size_t)NN * DD * 4, hipMemcpyDeviceToDevice, stream);
  k_heads<<<1, 128, 0, stream>>>(hbuf, batch, reg_W1, reg_b1, reg_W2, reg_b2,
                                 cls_W1, cls_b1, cls_W2, cls_b2, pen, (float*)d_out);
}
